// DecoderLayer_90374701842689
// MI455X (gfx1250) — matrix operand order repaired and checked by reading, compiled, not run on hardware
//
#include <hip/hip_runtime.h>
#include <cstdint>

#define BB   2
#define LL   2048
#define DM   512
#define NH   8
#define DH   64
#define DFF  2048
#define UU   409
#define ROWS (BB*LL)          // 4096
#define QTILE 16
#define UT   ((UU + QTILE - 1) / QTILE)   // 26 query tiles per (b,h)
#define KCH  64               // keys staged per chunk
#define KNCH 16               // kn_mean partial chunks

typedef __bf16 bf16;
typedef __attribute__((ext_vector_type(16))) __bf16 v16bf;
typedef __attribute__((ext_vector_type(8)))  float  v8f;

union V16U8 { v16bf v; uint32_t u[8]; };

__device__ __forceinline__ bf16 f2bf(float f) {
  union { float f; uint32_t u; } c; c.f = f;
  uint32_t r = (c.u + 0x7FFFu + ((c.u >> 16) & 1u)) >> 16;
  union { unsigned short s; bf16 b; } o; o.s = (unsigned short)r;
  return o.b;
}

__device__ __forceinline__ void load_frag_a(V16U8& f, const bf16* arow, int k0, int hg) {
#pragma unroll
  for (int j = 0; j < 8; ++j) {
    const int ka = ((j >> 2) << 4) + (hg << 3) + ((j & 3) << 1);
    f.u[j] = *(const uint32_t*)(arow + k0 + ka);
  }
}
__device__ __forceinline__ void load_frag_b(V16U8& f, const bf16* brow, int k0) {
#pragma unroll
  for (int j = 0; j < 8; ++j) f.u[j] = *(const uint32_t*)(brow + k0 + ((j >> 2) << 4) + ((j & 3) << 1));
}

// ---------------------------------------------------------------------------
// GEMM: C[M,N] = A[M,K](bf16) @ W[K,N] + bias, W given as WT[N,K](bf16).
// One wave computes a 32x64 tile: 8 accumulators, A fragments reused 4x,
// B fragments reused 2x. K loop is unrolled by 64 with ping-pong fragment
// buffers so loads of the next half-step overlap the current 8 WMMAs.
// Epilogue fuses bias, residual add, ReLU, f32/bf16 stores.
// ---------------------------------------------------------------------------
__global__ void __launch_bounds__(256)
gemm_bf16_wmma(const bf16* __restrict__ A, const bf16* __restrict__ WT,
               const float* __restrict__ bias, const float* __restrict__ res,
               float* __restrict__ Cf, bf16* __restrict__ Cb,
               int M, int N, int K, int do_relu)
{
  const int wave = (int)((blockIdx.x * blockDim.x + threadIdx.x) >> 5);
  const int lane = (int)(threadIdx.x & 31);
  const int strips = N >> 6;
  const int tm = wave / strips;            // 32-row tile
  const int tn = wave - tm * strips;       // 64-col strip
  if (tm * 32 >= M) return;                // wave-uniform exit

  const int mr = lane & 15;
  const int hg = lane >> 4;

  const bf16* arow0 = A + (size_t)(tm * 32 + mr) * K;
  const bf16* arow1 = arow0 + (size_t)16 * K;
  const bf16* bbase = WT + (size_t)(tn * 64 + mr) * K + (hg << 3);

  v8f acc[8] = {};
  V16U8 Af[2][2], Bf[2][4];                // [buffer][tile]

  load_frag_a(Af[0][0], arow0, 0, hg);
  load_frag_a(Af[0][1], arow1, 0, hg);
#pragma unroll
  for (int t = 0; t < 4; ++t) load_frag_b(Bf[0][t], bbase + (size_t)(t * 16) * K, 0);

  for (int k0 = 0; k0 < K; k0 += 64) {
    load_frag_a(Af[1][0], arow0, k0 + 32, hg);
    load_frag_a(Af[1][1], arow1, k0 + 32, hg);
#pragma unroll
    for (int t = 0; t < 4; ++t) load_frag_b(Bf[1][t], bbase + (size_t)(t * 16) * K, k0 + 32);
#pragma unroll
    for (int t = 0; t < 4; ++t) {
      acc[t]     = __builtin_amdgcn_wmma_f32_16x16x32_bf16(false, Af[0][0].v, false, Bf[0][t].v, (short)0, acc[t],     false, false);
      acc[4 + t] = __builtin_amdgcn_wmma_f32_16x16x32_bf16(false, Af[0][1].v, false, Bf[0][t].v, (short)0, acc[4 + t], false, false);
    }
    if (k0 + 64 < K) {
      load_frag_a(Af[0][0], arow0, k0 + 64, hg);
      load_frag_a(Af[0][1], arow1, k0 + 64, hg);
#pragma unroll
      for (int t = 0; t < 4; ++t) load_frag_b(Bf[0][t], bbase + (size_t)(t * 16) * K, k0 + 64);
    }
#pragma unroll
    for (int t = 0; t < 4; ++t) {
      acc[t]     = __builtin_amdgcn_wmma_f32_16x16x32_bf16(false, Af[1][0].v, false, Bf[1][t].v, (short)0, acc[t],     false, false);
      acc[4 + t] = __builtin_amdgcn_wmma_f32_16x16x32_bf16(false, Af[1][1].v, false, Bf[1][t].v, (short)0, acc[4 + t], false, false);
    }
  }

#pragma unroll
  for (int mt = 0; mt < 2; ++mt) {
#pragma unroll
    for (int t = 0; t < 4; ++t) {
#pragma unroll
      for (int j = 0; j < 8; ++j) {
        const int row = tm * 32 + mt * 16 + j + (hg << 3);
        const int col = tn * 64 + t * 16 + mr;
        float vv = acc[mt * 4 + t][j];
        if (bias) vv += bias[col];
        if (res)  vv += res[(size_t)row * N + col];
        if (do_relu) vv = vv > 0.f ? vv : 0.f;
        const size_t o = (size_t)row * N + col;
        if (Cf) Cf[o] = vv;
        if (Cb) Cb[o] = f2bf(vv);
      }
    }
  }
}

// ---------------------------------------------------------------------------
// LayerNorm over DM=512 per row -> bf16 (consumed only as WMMA A operand).
// ---------------------------------------------------------------------------
__global__ void __launch_bounds__(256)
layernorm_k(const float* __restrict__ x, const float* __restrict__ g,
            const float* __restrict__ bta, bf16* __restrict__ ob)
{
  __shared__ float sred[16];
  const int row = blockIdx.x;
  const float* xr = x + (size_t)row * DM;
  float s = 0.f, ss = 0.f;
  for (int i = threadIdx.x; i < DM; i += 256) { float v = xr[i]; s += v; ss += v * v; }
  for (int o = 16; o > 0; o >>= 1) { s += __shfl_xor(s, o, 32); ss += __shfl_xor(ss, o, 32); }
  if ((threadIdx.x & 31) == 0) { sred[threadIdx.x >> 5] = s; sred[8 + (threadIdx.x >> 5)] = ss; }
  __syncthreads();
  if (threadIdx.x == 0) {
    float ts = 0.f, tss = 0.f;
    for (int w = 0; w < 8; ++w) { ts += sred[w]; tss += sred[8 + w]; }
    sred[0] = ts; sred[8] = tss;
  }
  __syncthreads();
  const float mean = sred[0] / DM;
  const float var  = sred[8] / DM - mean * mean;
  const float inv  = rsqrtf(var + 1e-5f);
  for (int i = threadIdx.x; i < DM; i += 256)
    ob[(size_t)row * DM + i] = f2bf((xr[i] - mean) * inv * g[i] + bta[i]);
}

// W (K x N, f32, row-major) -> WT (N x K, bf16), LDS-tiled transpose.
__global__ void transpose_to_bf16(const float* __restrict__ W, bf16* __restrict__ WT,
                                  int K, int N)
{
  __shared__ float tile[32][33];
  const int kb = blockIdx.x * 32, nb = blockIdx.y * 32;
  for (int r = threadIdx.y; r < 32; r += 8)
    tile[r][threadIdx.x] = W[(size_t)(kb + r) * N + nb + threadIdx.x];
  __syncthreads();
  for (int r = threadIdx.y; r < 32; r += 8)
    WT[(size_t)(nb + r) * K + kb + threadIdx.x] = f2bf(tile[threadIdx.x][r]);
}

__global__ void cvt_bf16_k(const float* __restrict__ in, bf16* __restrict__ out, int n)
{
  int i = blockIdx.x * blockDim.x + threadIdx.x;
  if (i < n) out[i] = f2bf(in[i]);
}

__global__ void zero_u32_k(uint32_t* __restrict__ p, int n)
{
  int i = blockIdx.x * blockDim.x + threadIdx.x;
  if (i < n) p[i] = 0u;
}

// ---------------------------------------------------------------------------
// kn_mean[b,h,:] = mean_l( k-row / ||k-row|| ), split into KNCH partial chunks
// (256 concurrent waves) + deterministic tree reduce.
// mean_k(qn . kn) == qn . mean_k(kn)  -- removes the full LxL score matrix.
// ---------------------------------------------------------------------------
__global__ void __launch_bounds__(32)
knmean_part_k(const float* __restrict__ kmat, float* __restrict__ part)
{
  const int bh = blockIdx.x / KNCH, c = blockIdx.x % KNCH;
  const int b = bh / NH, h = bh % NH;
  const int lane = threadIdx.x;
  const int rows = LL / KNCH;
  const float* base = kmat + (size_t)b * LL * DM + h * DH;
  float a0 = 0.f, a1 = 0.f;
  for (int l = c * rows; l < (c + 1) * rows; ++l) {
    const float* r = base + (size_t)l * DM;
    float v0 = r[lane], v1 = r[lane + 32];
    float ss = v0 * v0 + v1 * v1;
    for (int o = 16; o > 0; o >>= 1) ss += __shfl_xor(ss, o, 32);
    float rn = rsqrtf(ss);
    a0 += v0 * rn; a1 += v1 * rn;
  }
  part[((size_t)bh * KNCH + c) * DH + lane]      = a0;
  part[((size_t)bh * KNCH + c) * DH + lane + 32] = a1;
}

__global__ void __launch_bounds__(64)
knmean_red_k(const float* __restrict__ part, float* __restrict__ knm)
{
  const int bh = blockIdx.x, d = threadIdx.x;
  float s = 0.f;
  for (int c = 0; c < KNCH; ++c) s += part[((size_t)bh * KNCH + c) * DH + d];
  knm[bh * DH + d] = s / LL;
}

// mean_scores[b,h,l] = (q . kn_mean) / ||q||.  One wave per query row.
__global__ void __launch_bounds__(256)
meanscore_k(const float* __restrict__ q, const float* __restrict__ knm,
            float* __restrict__ ms)
{
  const int gw = (int)((blockIdx.x * blockDim.x + threadIdx.x) >> 5);
  const int lane = (int)(threadIdx.x & 31);
  const int l = gw % LL, bh = gw / LL;
  const int b = bh / NH, h = bh % NH;
  const float* r = q + (size_t)(b * LL + l) * DM + h * DH;
  float v0 = r[lane], v1 = r[lane + 32];
  float ss = v0 * v0 + v1 * v1;
  float dp = v0 * knm[bh * DH + lane] + v1 * knm[bh * DH + lane + 32];
  for (int o = 16; o > 0; o >>= 1) { ss += __shfl_xor(ss, o, 32); dp += __shfl_xor(dp, o, 32); }
  if (lane == 0) ms[(size_t)bh * LL + l] = dp * rsqrtf(ss);
}

// Full bitonic sort of 2048 (value desc, index asc) in LDS; emit first UU ids.
__device__ __forceinline__ bool before_pair(float va, int ia, float vb, int ib)
{ return (va > vb) || (va == vb && ia < ib); }

__global__ void __launch_bounds__(256)
topk_k(const float* __restrict__ ms, int* __restrict__ tidx)
{
  __shared__ float sv[LL];
  __shared__ int   si[LL];
  const int bh = blockIdx.x;
  for (int i = threadIdx.x; i < LL; i += 256) { sv[i] = ms[(size_t)bh * LL + i]; si[i] = i; }
  __syncthreads();
  for (int kk = 2; kk <= LL; kk <<= 1) {
    for (int j = kk >> 1; j > 0; j >>= 1) {
      for (int i = threadIdx.x; i < LL; i += 256) {
        int ixj = i ^ j;
        if (ixj > i) {
          float va = sv[i], vb = sv[ixj]; int ia = si[i], ib = si[ixj];
          bool iFirst  = before_pair(va, ia, vb, ib);
          bool lowDesc = ((i & kk) == 0);
          if (lowDesc ? !iFirst : iFirst) { sv[i] = vb; sv[ixj] = va; si[i] = ib; si[ixj] = ia; }
        }
      }
      __syncthreads();
    }
  }
  for (int i = threadIdx.x; i < UU; i += 256) tidx[bh * UU + i] = si[i];
}

// ---------------------------------------------------------------------------
// Sparse attention, tiled: one block per (b,h, 16 selected queries).
// K/V are staged through LDS in 64-key chunks (rows padded to 65 floats so
// per-lane row reads hit distinct banks); softmax is online, accumulators in
// registers (each wave owns 2 queries, each lane 2 channels per query).
// L2 traffic: ~16x lower than one-block-per-query.
// ---------------------------------------------------------------------------
__global__ void __launch_bounds__(256)
attn_k(const float* __restrict__ q, const float* __restrict__ kmat,
       const float* __restrict__ vmat, const int* __restrict__ tidx,
       bf16* __restrict__ ctxb)
{
  __shared__ float qs[QTILE][DH];
  __shared__ float ks[KCH][DH + 1];
  __shared__ float vs[KCH][DH + 1];
  __shared__ float ps[QTILE][KCH];
  const int bh = blockIdx.x / UT;
  const int qt = blockIdx.x % UT;
  const int b = bh / NH, h = bh % NH;
  const int tid = threadIdx.x, lane = tid & 31, wv = tid >> 5;
  const int q0 = wv * 2, q1 = wv * 2 + 1;

  for (int i = tid; i < QTILE * DH; i += 256) {
    int qq = i >> 6, d = i & 63;
    int ui = qt * QTILE + qq;
    int qrow = tidx[bh * UU + (ui < UU ? ui : UU - 1)];
    qs[qq][d] = q[(size_t)(b * LL + qrow) * DM + h * DH + d];
  }

  float m0 = -3.4e38f, m1 = -3.4e38f, l0 = 0.f, l1 = 0.f;
  float a0a = 0.f, a0b = 0.f, a1a = 0.f, a1b = 0.f;

  for (int c0 = 0; c0 < LL; c0 += KCH) {
    __syncthreads();
    for (int i = tid; i < KCH * DH; i += 256) {
      int kk = i >> 6, d = i & 63;
      size_t gofs = (size_t)(b * LL + c0 + kk) * DM + h * DH + d;
      ks[kk][d] = kmat[gofs];
      vs[kk][d] = vmat[gofs];
    }
    __syncthreads();

    // scores: lane handles keys (lane, lane+32) for this wave's two queries
    float s00 = 0.f, s01 = 0.f, s10 = 0.f, s11 = 0.f;
    for (int d = 0; d < DH; ++d) {
      float k0v = ks[lane][d], k1v = ks[lane + 32][d];
      float q0v = qs[q0][d],   q1v = qs[q1][d];
      s00 += q0v * k0v; s01 += q0v * k1v;
      s10 += q1v * k0v; s11 += q1v * k1v;
    }
    s00 *= 0.125f; s01 *= 0.125f; s10 *= 0.125f; s11 *= 0.125f;   // 1/sqrt(64)

    float cm0 = fmaxf(s00, s01), cm1 = fmaxf(s10, s11);
    for (int o = 16; o > 0; o >>= 1) {
      cm0 = fmaxf(cm0, __shfl_xor(cm0, o, 32));
      cm1 = fmaxf(cm1, __shfl_xor(cm1, o, 32));
    }
    float nm0 = fmaxf(m0, cm0), nm1 = fmaxf(m1, cm1);
    float f0 = __expf(m0 - nm0), f1 = __expf(m1 - nm1);
    float e00 = __expf(s00 - nm0), e01 = __expf(s01 - nm0);
    float e10 = __expf(s10 - nm1), e11 = __expf(s11 - nm1);
    ps[q0][lane] = e00; ps[q0][lane + 32] = e01;
    ps[q1][lane] = e10; ps[q1][lane + 32] = e11;
    float es0 = e00 + e01, es1 = e10 + e11;
    for (int o = 16; o > 0; o >>= 1) {
      es0 += __shfl_xor(es0, o, 32);
      es1 += __shfl_xor(es1, o, 32);
    }
    l0 = l0 * f0 + es0; l1 = l1 * f1 + es1;
    m0 = nm0; m1 = nm1;
    a0a *= f0; a0b *= f0; a1a *= f1; a1b *= f1;

    // P@V: lane owns channels (lane, lane+32); ps reads are wave-local
    for (int kk = 0; kk < KCH; ++kk) {
      float p0 = ps[q0][kk], p1 = ps[q1][kk];
      float v0 = vs[kk][lane], v1 = vs[kk][lane + 32];
      a0a += p0 * v0; a0b += p0 * v1;
      a1a += p1 * v0; a1b += p1 * v1;
    }
  }

  const int ui0 = qt * QTILE + q0, ui1 = qt * QTILE + q1;
  if (ui0 < UU) {
    int qrow = tidx[bh * UU + ui0];
    size_t o = (size_t)(b * LL + qrow) * DM + h * DH;
    float inv = 1.f / l0;
    ctxb[o + lane]      = f2bf(a0a * inv);
    ctxb[o + lane + 32] = f2bf(a0b * inv);
  }
  if (ui1 < UU) {
    int qrow = tidx[bh * UU + ui1];
    size_t o = (size_t)(b * LL + qrow) * DM + h * DH;
    float inv = 1.f / l1;
    ctxb[o + lane]      = f2bf(a1a * inv);
    ctxb[o + lane + 32] = f2bf(a1b * inv);
  }
}

// ---------------------------------------------------------------------------
extern "C" void kernel_launch(void* const* d_in, const int* in_sizes, int n_in,
                              void* d_out, int out_size, void* d_ws, size_t ws_size,
                              hipStream_t stream)
{
  (void)in_sizes; (void)n_in; (void)out_size; (void)ws_size;
  const float* x_in = (const float*)d_in[0];
  const float* enc  = (const float*)d_in[1];
  const float* sa_w[4] = { (const float*)d_in[2], (const float*)d_in[4],
                           (const float*)d_in[6], (const float*)d_in[8] };
  const float* sa_b[4] = { (const float*)d_in[3], (const float*)d_in[5],
                           (const float*)d_in[7], (const float*)d_in[9] };
  const float* ca_w[4] = { (const float*)d_in[10], (const float*)d_in[12],
                           (const float*)d_in[14], (const float*)d_in[16] };
  const float* ca_b[4] = { (const float*)d_in[11], (const float*)d_in[13],
                           (const float*)d_in[15], (const float*)d_in[17] };
  const float* ffw1 = (const float*)d_in[18]; const float* ffb1 = (const float*)d_in[19];
  const float* ffw2 = (const float*)d_in[20]; const float* ffb2 = (const float*)d_in[21];
  const float* lng[3] = { (const float*)d_in[22], (const float*)d_in[24], (const float*)d_in[26] };
  const float* lnb[3] = { (const float*)d_in[23], (const float*)d_in[25], (const float*)d_in[27] };

  char* base = (char*)d_ws; size_t off = 0;
  auto alloc = [&](size_t bytes) -> void* {
    void* p = base + off; off = (off + bytes + 255) & ~(size_t)255; return p;
  };

  bf16* wtT[10];
  for (int i = 0; i < 8; ++i) wtT[i] = (bf16*)alloc((size_t)DM * DM * 2);
  wtT[8] = (bf16*)alloc((size_t)DFF * DM * 2);   // w1^T : N=DFF rows, K=DM
  wtT[9] = (bf16*)alloc((size_t)DM * DFF * 2);   // w2^T : N=DM rows,  K=DFF
  bf16*  encb = (bf16*) alloc((size_t)ROWS * DM * 2);
  float* xw   = (float*)alloc((size_t)ROWS * DM * 4);
  bf16*  xnb  = (bf16*) alloc((size_t)ROWS * DM * 2);
  float* qf   = (float*)alloc((size_t)ROWS * DM * 4);
  float* kf   = (float*)alloc((size_t)ROWS * DM * 4);
  float* vf   = (float*)alloc((size_t)ROWS * DM * 4);
  float* knmp = (float*)alloc((size_t)BB * NH * KNCH * DH * 4);
  float* knm  = (float*)alloc((size_t)BB * NH * DH * 4);
  float* ms   = (float*)alloc((size_t)BB * NH * LL * 4);
  int*   tix  = (int*)  alloc((size_t)BB * NH * UU * 4);
  bf16*  ctxb = (bf16*) alloc((size_t)ROWS * DM * 2);
  bf16*  hb   = (bf16*) alloc((size_t)ROWS * DFF * 2);

  auto tpose = [&](const float* W, bf16* WTp, int K, int N) {
    dim3 g(K / 32, N / 32), blk(32, 8);
    transpose_to_bf16<<<g, blk, 0, stream>>>(W, WTp, K, N);
  };
  for (int i = 0; i < 4; ++i) tpose(sa_w[i], wtT[i],     DM, DM);
  for (int i = 0; i < 4; ++i) tpose(ca_w[i], wtT[4 + i], DM, DM);
  tpose(ffw1, wtT[8], DM, DFF);
  tpose(ffw2, wtT[9], DFF, DM);
  { int n = ROWS * DM; cvt_bf16_k<<<(n + 255) / 256, 256, 0, stream>>>(enc, encb, n); }
  hipMemcpyAsync(xw, x_in, (size_t)ROWS * DM * 4, hipMemcpyDeviceToDevice, stream);

  auto gemm = [&](const bf16* A, const bf16* WTp, const float* bias, const float* res,
                  float* Cf, bf16* Cb, int M, int N, int K, int relu) {
    int waves = (M / 32) * (N / 64);
    gemm_bf16_wmma<<<waves / 8, 256, 0, stream>>>(A, WTp, bias, res, Cf, Cb, M, N, K, relu);
  };

  auto attn_block = [&](const bf16* qinb, const bf16* kvinb,
                        bf16* const* wT, const float* const* bias, float* xio) {
    gemm(qinb,  wT[0], bias[0], nullptr, qf, nullptr, ROWS, DM, DM, 0);
    gemm(kvinb, wT[1], bias[1], nullptr, kf, nullptr, ROWS, DM, DM, 0);
    gemm(kvinb, wT[2], bias[2], nullptr, vf, nullptr, ROWS, DM, DM, 0);
    knmean_part_k<<<BB * NH * KNCH, 32, 0, stream>>>(kf, knmp);
    knmean_red_k<<<BB * NH, 64, 0, stream>>>(knmp, knm);
    meanscore_k<<<(BB * NH * LL) / 8, 256, 0, stream>>>(qf, knm, ms);
    topk_k<<<BB * NH, 256, 0, stream>>>(ms, tix);
    { int n = ROWS * DM / 2; zero_u32_k<<<(n + 255) / 256, 256, 0, stream>>>((uint32_t*)ctxb, n); }
    attn_k<<<BB * NH * UT, 256, 0, stream>>>(qf, kf, vf, tix, ctxb);
    gemm(ctxb, wT[3], bias[3], xio, xio, nullptr, ROWS, DM, DM, 0);   // + residual
  };

  // self-attention
  layernorm_k<<<ROWS, 256, 0, stream>>>(xw, lng[0], lnb[0], xnb);
  { bf16* wT[4] = { wtT[0], wtT[1], wtT[2], wtT[3] };
    attn_block(xnb, xnb, wT, sa_b, xw); }
  // cross-attention
  layernorm_k<<<ROWS, 256, 0, stream>>>(xw, lng[1], lnb[1], xnb);
  { bf16* wT[4] = { wtT[4], wtT[5], wtT[6], wtT[7] };
    attn_block(xnb, encb, wT, ca_b, xw); }
  // FFN
  layernorm_k<<<ROWS, 256, 0, stream>>>(xw, lng[2], lnb[2], xnb);
  gemm(xnb, wtT[8], ffb1, nullptr, nullptr, hb, ROWS, DFF, DM, 1);          // ReLU -> bf16
  gemm(hb,  wtT[9], ffb2, xw, (float*)d_out, nullptr, ROWS, DM, DFF, 0);    // + residual -> out
}
